// ECPEC_37984690766363
// MI455X (gfx1250) — matrix operand order repaired and checked by reading, not compiled, not run on hardware
//
#include <hip/hip_runtime.h>
#include <cmath>

typedef __attribute__((ext_vector_type(16))) __bf16 bf16x16;
typedef __attribute__((ext_vector_type(8)))  float  floatx8;
typedef int v4i_vec __attribute__((__vector_size__(16)));

#define AS1 __attribute__((address_space(1)))
#define AS3 __attribute__((address_space(3)))

#define BB 256
#define TT 128
#define DD 1024
#define HH 512
#define NE 64
#define NCC 128
#define PP 50
#define K2  3123   /* 3D+1+P  (logical) */
#define K2P 3136   /* padded to mult of 32 */
#define K3  3173   /* 3D+51+P (logical) */
#define K3P 3200   /* padded to mult of 32 */
#define LSLOPE 0.01f
#define BNEPS  1e-5f

// ---------------------------------------------------------------------------
// CDNA5 async global->LDS copy (ASYNCcnt path) with safe fallbacks
// ---------------------------------------------------------------------------
__device__ __forceinline__ void async_copy_b128(void* lds_dst, const void* gsrc) {
#if defined(__HIP_DEVICE_COMPILE__)
#if __has_builtin(__builtin_amdgcn_global_load_async_to_lds_b128)
    __builtin_amdgcn_global_load_async_to_lds_b128(
        (AS1 v4i_vec*)gsrc, (AS3 v4i_vec*)lds_dst, 0, 0);
#else
    asm volatile("global_load_async_to_lds_b128 %0, %1, off"
                 :: "v"((unsigned)(unsigned long long)(AS3 char*)lds_dst),
                    "v"(gsrc)
                 : "memory");
#endif
#else
    (void)lds_dst; (void)gsrc;
#endif
}

__device__ __forceinline__ void async_wait0() {
#if defined(__HIP_DEVICE_COMPILE__)
#if __has_builtin(__builtin_amdgcn_s_wait_asynccnt)
    __builtin_amdgcn_s_wait_asynccnt(0);
#else
    asm volatile("s_wait_asynccnt 0x0" ::: "memory");
#endif
#endif
}

// ---------------------------------------------------------------------------
// Block reduction helper (256 threads)
// ---------------------------------------------------------------------------
__device__ __forceinline__ float block_reduce_256(float v, float* sbuf) {
    int tid = threadIdx.x;
    sbuf[tid] = v;
    __syncthreads();
    for (int s = 128; s > 0; s >>= 1) {
        if (tid < s) sbuf[tid] += sbuf[tid + s];
        __syncthreads();
    }
    float r = sbuf[0];
    __syncthreads();
    return r;
}

// ---------------------------------------------------------------------------
// f32 -> bf16 conversion with optional row padding (dst cols >= src cols)
// ---------------------------------------------------------------------------
__global__ __launch_bounds__(256)
void conv_bf16(const float* __restrict__ src, __bf16* __restrict__ dst,
               long rows, long scols, long dcols)
{
    long total = rows * dcols;
    for (long i = (long)blockIdx.x * 256 + threadIdx.x; i < total;
         i += (long)gridDim.x * 256) {
        long r = i / dcols, c = i - r * dcols;
        dst[i] = (c < scols) ? (__bf16)src[r * scols + c] : (__bf16)0.0f;
    }
}

// ---------------------------------------------------------------------------
// WMMA GEMM: C[M,N] = A[M,K]*W[N,K]^T (+bias0+bias1) (+=C), A/W bf16, C f32.
// K must be a multiple of 32; M,N multiples of 128.
// 128x128 tile/WG, 8 waves (4x2), async double-buffered LDS staging.
// A rows optionally gathered through the length-reversal map (backward LSTM).
// ---------------------------------------------------------------------------
#define TM 128
#define TN 128
#define TK 32
#define LDSS 40   /* 80B rows: 16B-aligned chunks, spread banks */

__global__ __launch_bounds__(256)
void gemm_bf16_wmma(const __bf16* __restrict__ A,
                    const __bf16* __restrict__ W,
                    float* __restrict__ C,
                    int M, int N, int K,
                    long lda, long ldw, long ldc,
                    const float* __restrict__ bias0,
                    const float* __restrict__ bias1,
                    int accumulate,
                    const int* __restrict__ lengths, int T_seq)
{
    __shared__ __bf16 As[2][TM][LDSS];
    __shared__ __bf16 Bs[2][TN][LDSS];

    const int tid  = threadIdx.x;
    const int lane = tid & 31;
    const int wave = tid >> 5;     // 0..7
    const int wm   = wave >> 1;    // 0..3  (M)
    const int wn   = wave & 1;     // 0..1  (N)
    const int half = lane >> 4;    // K-phase selector
    const int l16  = lane & 15;

    const int block_m = blockIdx.y * TM;
    const int block_n = blockIdx.x * TN;

    floatx8 acc[2][4];
    #pragma unroll
    for (int mt = 0; mt < 2; ++mt)
        #pragma unroll
        for (int nt = 0; nt < 4; ++nt)
            acc[mt][nt] = (floatx8)0.0f;

    // stage one 128x32 bf16 tile pair (A and W) into LDS buffer `buf`
    auto stage = [&](int k0, int buf) {
        for (int i = tid; i < TM * 4; i += 256) {       // 4 x 16B chunks / row
            int r = i >> 2, ch = (i & 3) * 8;
            int gr = block_m + r;
            long src_row;
            if (lengths) {
                int b = gr / T_seq, t = gr - b * T_seq;
                int len = lengths[b];
                int st = (t < len) ? (len - 1 - t) : t;
                src_row = (long)b * T_seq + st;
            } else {
                src_row = gr;
            }
            async_copy_b128(&As[buf][r][ch], A + src_row * lda + k0 + ch);
        }
        for (int i = tid; i < TN * 4; i += 256) {
            int n = i >> 2, ch = (i & 3) * 8;
            async_copy_b128(&Bs[buf][n][ch], W + (long)(block_n + n) * ldw + k0 + ch);
        }
    };

    const int nk = K / TK;
    stage(0, 0);

    for (int kb = 0; kb < nk; ++kb) {
        const int cur = kb & 1;
        async_wait0();        // own-wave async copies done
        __syncthreads();      // all waves' copies of tile kb visible; kb-1 compute done
        if (kb + 1 < nk) stage((kb + 1) * TK, cur ^ 1);

        bf16x16 bfrag[4];
        #pragma unroll
        for (int nt = 0; nt < 4; ++nt) {
            int n  = wn * 64 + nt * 16 + l16;
            int kb16 = half * 8;
            #pragma unroll
            for (int i = 0; i < 16; ++i) bfrag[nt][i] = Bs[cur][n][kb16 + (i & 7) + ((i >> 3) << 4)];
        }
        #pragma unroll
        for (int mt = 0; mt < 2; ++mt) {
            int m  = wm * 32 + mt * 16 + l16;
            int ka = half * 8;
            bf16x16 afrag;
            #pragma unroll
            for (int i = 0; i < 8; ++i) afrag[i]     = As[cur][m][ka + i];
            #pragma unroll
            for (int i = 0; i < 8; ++i) afrag[8 + i] = As[cur][m][ka + 16 + i];
            #pragma unroll
            for (int nt = 0; nt < 4; ++nt) {
                acc[mt][nt] = __builtin_amdgcn_wmma_f32_16x16x32_bf16(
                    false, afrag, false, bfrag[nt],
                    (short)0, acc[mt][nt], false, false);
            }
        }
    }

    #pragma unroll
    for (int mt = 0; mt < 2; ++mt) {
        #pragma unroll
        for (int nt = 0; nt < 4; ++nt) {
            int gn = block_n + wn * 64 + nt * 16 + l16;
            float bsum = 0.0f;
            if (bias0) bsum += bias0[gn];
            if (bias1) bsum += bias1[gn];
            #pragma unroll
            for (int r = 0; r < 8; ++r) {
                int gm = block_m + wm * 32 + mt * 16 + half * 8 + r;
                long idx = (long)gm * ldc + gn;
                float v = acc[mt][nt][r] + bsum;
                if (accumulate) C[idx] += v; else C[idx] = v;
            }
        }
    }
}

// ---------------------------------------------------------------------------
__global__ void zero_f(float* __restrict__ p, long n) {
    long i = (long)blockIdx.x * blockDim.x + threadIdx.x;
    if (i < n) p[i] = 0.0f;
}

// ---------------------------------------------------------------------------
// LSTM cell (one time step); writes f32 state + bf16 mirrors for the GEMMs
// ---------------------------------------------------------------------------
__global__ __launch_bounds__(256)
void lstm_cell(const float* __restrict__ gates,
               float* __restrict__ h, float* __restrict__ c,
               __bf16* __restrict__ hB,
               float* __restrict__ Hseq, __bf16* __restrict__ HseqB,
               const int* __restrict__ lengths,
               int t, int reverse, int col_off)
{
    int idx = blockIdx.x * blockDim.x + threadIdx.x;   // b*HH + j
    int b = idx >> 9;
    int j = idx & (HH - 1);
    long grow = ((long)b * TT + t) * (4 * HH);
    float gi = gates[grow + j];
    float gf = gates[grow + HH + j];
    float gg = gates[grow + 2 * HH + j];
    float go = gates[grow + 3 * HH + j];
    float cv = c[idx], hv = h[idx];
    float i_s = 1.0f / (1.0f + expf(-gi));
    float f_s = 1.0f / (1.0f + expf(-gf));
    float o_s = 1.0f / (1.0f + expf(-go));
    float c_new = f_s * cv + i_s * tanhf(gg);
    float h_new = o_s * tanhf(c_new);
    int len = lengths[b];
    bool m = (t < len);
    float hout = m ? h_new : hv;
    c[idx] = m ? c_new : cv;
    h[idx] = hout;
    hB[idx] = (__bf16)hout;
    int t_out = reverse ? (m ? (len - 1 - t) : t) : t;
    float ov = m ? h_new : 0.0f;
    long oidx = ((long)b * TT + t_out) * (2 * HH) + col_off + j;
    Hseq[oidx]  = ov;
    HseqB[oidx] = (__bf16)ov;
}

// ---------------------------------------------------------------------------
__global__ __launch_bounds__(256)
void colstats(const float* __restrict__ Z, int rows, int cols,
              float* __restrict__ mean, float* __restrict__ var)
{
    __shared__ float red[256];
    int cidx = blockIdx.x;
    float s = 0.0f, s2 = 0.0f;
    for (int r = threadIdx.x; r < rows; r += 256) {
        float v = Z[(long)r * cols + cidx];
        s += v; s2 += v * v;
    }
    float ts  = block_reduce_256(s, red);
    float ts2 = block_reduce_256(s2, red);
    if (threadIdx.x == 0) {
        float m = ts / rows;
        mean[cidx] = m;
        var[cidx]  = ts2 / rows - m * m;
    }
}

// ---------------------------------------------------------------------------
__global__ __launch_bounds__(256)
void alpha_kernel(const float* __restrict__ Z,
                  const float* __restrict__ mean, const float* __restrict__ var,
                  const float* __restrict__ s2w, float* __restrict__ alpha)
{
    __shared__ float red[256];
    int row = blockIdx.x;
    float p = 0.0f;
    for (int d = threadIdx.x; d < DD; d += 256) {
        float v = (Z[(long)row * DD + d] - mean[d]) * rsqrtf(var[d] + BNEPS);
        v = (v >= 0.0f) ? v : LSLOPE * v;
        p += v * s2w[d];
    }
    float tot = block_reduce_256(p, red);
    if (threadIdx.x == 0) alpha[row] = tot;
}

// ---------------------------------------------------------------------------
__global__ __launch_bounds__(256)
void att_pool(const float* __restrict__ Hseq, const float* __restrict__ alpha,
              const int* __restrict__ lengths, float* __restrict__ U)
{
    __shared__ float red[256];
    __shared__ float am[TT];
    __shared__ float sinv;
    int b = blockIdx.x;
    int len = lengths[b];
    float part = 0.0f;
    for (int t = threadIdx.x; t < TT; t += 256) {
        float a = (t < len) ? alpha[b * TT + t] : 0.0f;
        am[t] = a; part += a;
    }
    float s = block_reduce_256(part, red);
    if (threadIdx.x == 0) sinv = 1.0f / (s + 1e-9f);
    __syncthreads();
    for (int d = threadIdx.x; d < DD; d += 256) {
        float acc = 0.0f;
        for (int t = 0; t < TT; ++t)
            acc += am[t] * Hseq[((long)b * TT + t) * DD + d];
        U[(long)b * DD + d] = acc * sinv;
    }
}

// ---------------------------------------------------------------------------
__global__ __launch_bounds__(256)
void chunk_att(const float* __restrict__ U, float* __restrict__ chunkEmb)
{
    __shared__ float red[256];
    __shared__ float sc[16];
    int e = blockIdx.x >> 3, n = blockIdx.x & 7;
    const float* emo = U + (long)e * DD;
    const float* cau = U + (long)(NE + n * 16) * DD;
    for (int ci = 0; ci < 16; ++ci) {
        float p = 0.0f;
        for (int d = threadIdx.x; d < DD; d += 256)
            p += emo[d] * cau[(long)ci * DD + d];
        float tot = block_reduce_256(p, red);
        if (threadIdx.x == 0) sc[ci] = tot;
        __syncthreads();
    }
    if (threadIdx.x == 0) {
        float mx = sc[0];
        for (int ci = 1; ci < 16; ++ci) mx = fmaxf(mx, sc[ci]);
        float s = 0.0f;
        for (int ci = 0; ci < 16; ++ci) { sc[ci] = expf(sc[ci] - mx); s += sc[ci]; }
        float inv = 1.0f / s;
        for (int ci = 0; ci < 16; ++ci) sc[ci] *= inv;
    }
    __syncthreads();
    for (int d = threadIdx.x; d < DD; d += 256) {
        float a = 0.0f;
        for (int ci = 0; ci < 16; ++ci) a += sc[ci] * cau[(long)ci * DD + d];
        chunkEmb[(long)blockIdx.x * DD + d] = a;
    }
}

// ---------------------------------------------------------------------------
// delta2 row (bf16, padded): [emo, chunkEmb, dist, emo*chunkEmb, pos(50), 0..]
// ---------------------------------------------------------------------------
__global__ __launch_bounds__(256)
void build_delta2(const float* __restrict__ U, const float* __restrict__ chunkEmb,
                  const float* __restrict__ pos, __bf16* __restrict__ delta2)
{
    __shared__ float red[256];
    int row = blockIdx.x;                 // e*8+n
    int tid = threadIdx.x;
    const float* emo = U + (long)(row >> 3) * DD;
    const float* y   = chunkEmb + (long)row * DD;
    float p = 0.0f;
    for (int d = tid; d < DD; d += 256) {
        float df = emo[d] - y[d]; p += df * df;
    }
    float dist = sqrtf(block_reduce_256(p, red));
    __bf16* dr = delta2 + (long)row * K2P;
    for (int d = tid; d < DD; d += 256) {
        dr[d]              = (__bf16)emo[d];
        dr[DD + d]         = (__bf16)y[d];
        dr[2 * DD + 1 + d] = (__bf16)(emo[d] * y[d]);
    }
    if (tid == 0) dr[2 * DD] = (__bf16)dist;
    if (tid < PP) dr[3 * DD + 1 + tid] = (__bf16)pos[(long)row * PP + tid];
    if (tid < (K2P - K2)) dr[K2 + tid] = (__bf16)0.0f;
}

// ---------------------------------------------------------------------------
// delta3 row (bf16, padded): [emo, cau, dist, emo*cau, dis(50), extra(50), 0..]
// ---------------------------------------------------------------------------
__global__ __launch_bounds__(256)
void build_delta3(const float* __restrict__ U, const float* __restrict__ dis,
                  const float* __restrict__ extra, __bf16* __restrict__ delta3)
{
    __shared__ float red[256];
    int row = blockIdx.x;                 // e*128 + c
    int tid = threadIdx.x;
    int e = row >> 7, cidx = row & 127;
    const float* emo = U + (long)e * DD;
    const float* y   = U + (long)(NE + cidx) * DD;
    float p = 0.0f;
    for (int d = tid; d < DD; d += 256) {
        float df = emo[d] - y[d]; p += df * df;
    }
    float dist = sqrtf(block_reduce_256(p, red));
    float ex = extra[e * 8 + (cidx >> 4)];
    __bf16* dr = delta3 + (long)row * K3P;
    for (int d = tid; d < DD; d += 256) {
        dr[d]              = (__bf16)emo[d];
        dr[DD + d]         = (__bf16)y[d];
        dr[2 * DD + 1 + d] = (__bf16)(emo[d] * y[d]);
    }
    if (tid == 0) dr[2 * DD] = (__bf16)dist;
    if (tid < PP) {
        dr[3 * DD + 1 + tid]  = (__bf16)dis[((long)e * NCC + cidx) * PP + tid];
        dr[3 * DD + 51 + tid] = (__bf16)ex;
    }
    if (tid < (K3P - K3)) dr[K3 + tid] = (__bf16)0.0f;
}

// ---------------------------------------------------------------------------
__global__ __launch_bounds__(256)
void bn_leaky_group(float* __restrict__ X, int rows_per_grp)
{
    int e = blockIdx.x;
    float* base = X + (long)e * rows_per_grp * DD;
    float inv_n = 1.0f / rows_per_grp;
    for (int d = threadIdx.x; d < DD; d += 256) {
        float s = 0.0f, s2 = 0.0f;
        for (int r = 0; r < rows_per_grp; ++r) {
            float v = base[(long)r * DD + d];
            s += v; s2 += v * v;
        }
        float m = s * inv_n;
        float var = s2 * inv_n - m * m;
        float inv = rsqrtf(var + BNEPS);
        for (int r = 0; r < rows_per_grp; ++r) {
            float v = (base[(long)r * DD + d] - m) * inv;
            base[(long)r * DD + d] = (v >= 0.0f) ? v : LSLOPE * v;
        }
    }
}

// ---------------------------------------------------------------------------
__global__ __launch_bounds__(256)
void head2_kernel(const float* __restrict__ hidden, const float* __restrict__ w,
                  const float* __restrict__ bias, float* __restrict__ out,
                  float* __restrict__ extra)
{
    __shared__ float red[256];
    int row = blockIdx.x;
    float p0 = 0.0f, p1 = 0.0f;
    for (int d = threadIdx.x; d < DD; d += 256) {
        float h = hidden[(long)row * DD + d];
        p0 += h * w[d];
        p1 += h * w[DD + d];
    }
    float z0 = block_reduce_256(p0, red);
    float z1 = block_reduce_256(p1, red);
    if (threadIdx.x == 0) {
        z0 += bias[0]; z1 += bias[1];
        float mx = fmaxf(z0, z1);
        float lse = mx + logf(expf(z0 - mx) + expf(z1 - mx));
        out[(long)row * 2]     = z0 - lse;
        out[(long)row * 2 + 1] = z1 - lse;
        if (extra) extra[row] = (z1 > z0) ? 1.0f : 0.0f;
    }
}

// ---------------------------------------------------------------------------
__global__ void setL(float* __restrict__ L, const int* __restrict__ label3)
{
    int e = threadIdx.x;
    if (e < NE) {
        for (int j = 0; j < 3; ++j) {
            int c = label3[e * 3 + j];
            if (c >= 0) L[e * NCC + c] = 1.0f;
        }
    }
}

// ---------------------------------------------------------------------------
// launcher
// ---------------------------------------------------------------------------
extern "C" void kernel_launch(void* const* d_in, const int* in_sizes, int n_in,
                              void* d_out, int out_size, void* d_ws, size_t ws_size,
                              hipStream_t stream)
{
    const float* word  = (const float*)d_in[0];
    const float* pos   = (const float*)d_in[1];
    const float* dis   = (const float*)d_in[2];
    const int*   slen  = (const int*)d_in[3];
    const int*   lab3  = (const int*)d_in[4];
    const float* Wih_f = (const float*)d_in[7];
    const float* Whh_f = (const float*)d_in[8];
    const float* bih_f = (const float*)d_in[9];
    const float* bhh_f = (const float*)d_in[10];
    const float* Wih_b = (const float*)d_in[11];
    const float* Whh_b = (const float*)d_in[12];
    const float* bih_b = (const float*)d_in[13];
    const float* bhh_b = (const float*)d_in[14];
    const float* s1_w  = (const float*)d_in[15];
    const float* s1_b  = (const float*)d_in[16];
    const float* s2_w  = (const float*)d_in[17];
    const float* W2_w  = (const float*)d_in[18];
    const float* W2_b  = (const float*)d_in[19];
    const float* Wo_w  = (const float*)d_in[20];
    const float* Wo_b  = (const float*)d_in[21];
    const float* W3_w  = (const float*)d_in[22];
    const float* W3_b  = (const float*)d_in[23];
    const float* cls_w = (const float*)d_in[24];
    const float* cls_b = (const float*)d_in[25];
    float* out = (float*)d_out;

    // ---- workspace layout (byte offsets, 256B aligned) ----
    char* base = (char*)d_ws;
    size_t off = 0;
    auto alloc = [&](size_t bytes) -> char* {
        char* p = base + off;
        off = (off + bytes + 255) & ~(size_t)255;
        return p;
    };
    const long MBT = (long)BB * TT;                       // 32768
    float*  XG    = (float*) alloc(MBT * 4 * HH * 4);     // 268MB (reused below)
    float*  HSEQ  = (float*) alloc(MBT * DD * 4);         // 134MB
    __bf16* HSEQB = (__bf16*)alloc(MBT * DD * 2);         // 67MB
    __bf16* WORDB = (__bf16*)alloc(MBT * DD * 2);         // 67MB
    __bf16* WIHF  = (__bf16*)alloc((long)4 * HH * DD * 2);
    __bf16* WIHB  = (__bf16*)alloc((long)4 * HH * DD * 2);
    __bf16* WHHF  = (__bf16*)alloc((long)4 * HH * HH * 2);
    __bf16* WHHB  = (__bf16*)alloc((long)4 * HH * HH * 2);
    __bf16* S1WB  = (__bf16*)alloc((long)DD * DD * 2);
    __bf16* W2WB  = (__bf16*)alloc((long)DD * K2P * 2);
    __bf16* W3WB  = (__bf16*)alloc((long)DD * K3P * 2);
    float*  HVEC  = (float*) alloc((long)BB * HH * 4);
    float*  CVEC  = (float*) alloc((long)BB * HH * 4);
    __bf16* HVECB = (__bf16*)alloc((long)BB * HH * 2);
    float*  MEANB = (float*) alloc(DD * 4);
    float*  VARB  = (float*) alloc(DD * 4);
    float*  ALPHA = (float*) alloc(MBT * 4);
    float*  UB    = (float*) alloc((long)BB * DD * 4);
    float*  CHUNK = (float*) alloc((long)NE * 8 * DD * 4);
    __bf16* D2    = (__bf16*)alloc((long)NE * 8 * K2P * 2);
    float*  HID2  = (float*) alloc((long)NE * 8 * DD * 4);
    float*  EXTRA = (float*) alloc(NE * 8 * 4);
    // reused regions inside XG (each earlier tenant is dead before reuse)
    float*  Z     = XG;                                   // 134MB
    __bf16* D3    = (__bf16*)XG;                          // 52MB
    float*  HID3  = (float*)((char*)XG + (size_t)MBT * DD * 4); // +134MB, 33.5MB

    // ---- one-time bf16 conversions (halve all GEMM stream traffic) ----
    conv_bf16<<<4096, 256, 0, stream>>>(word,  WORDB, MBT, DD, DD);
    conv_bf16<<<1024, 256, 0, stream>>>(Wih_f, WIHF, 4 * HH, DD, DD);
    conv_bf16<<<1024, 256, 0, stream>>>(Wih_b, WIHB, 4 * HH, DD, DD);
    conv_bf16<<<512,  256, 0, stream>>>(Whh_f, WHHF, 4 * HH, HH, HH);
    conv_bf16<<<512,  256, 0, stream>>>(Whh_b, WHHB, 4 * HH, HH, HH);
    conv_bf16<<<512,  256, 0, stream>>>(s1_w,  S1WB, DD, DD, DD);
    conv_bf16<<<1024, 256, 0, stream>>>(W2_w,  W2WB, DD, K2, K2P);
    conv_bf16<<<1024, 256, 0, stream>>>(W3_w,  W3WB, DD, K3, K3P);

    const long ld4H = (long)TT * 4 * HH;                  // xg row stride per b

    for (int dir = 0; dir < 2; ++dir) {
        const __bf16* Wih = dir ? WIHB : WIHF;
        const __bf16* Whh = dir ? WHHB : WHHF;
        const float*  bih = dir ? bih_b : bih_f;
        const float*  bhh = dir ? bhh_b : bhh_f;
        // xg = x(+rev) @ Wih^T + bih + bhh   [32768, 2048]
        gemm_bf16_wmma<<<dim3(16, 256), 256, 0, stream>>>(
            WORDB, Wih, XG, (int)MBT, 4 * HH, DD,
            DD, DD, 4 * HH, bih, bhh, 0,
            dir ? slen : nullptr, TT);
        zero_f<<<dim3((2 * BB * HH + 255) / 256), 256, 0, stream>>>(HVEC, 2L * BB * HH);
        zero_f<<<dim3((BB * HH / 2 + 255) / 256), 256, 0, stream>>>((float*)HVECB, BB * HH / 2);
        for (int t = 0; t < TT; ++t) {
            gemm_bf16_wmma<<<dim3(16, 2), 256, 0, stream>>>(
                HVECB, Whh, XG + (long)t * 4 * HH, BB, 4 * HH, HH,
                HH, HH, ld4H, nullptr, nullptr, 1, nullptr, 0);
            lstm_cell<<<dim3((BB * HH) / 256), 256, 0, stream>>>(
                XG, HVEC, CVEC, HVECB, HSEQ, HSEQB, slen, t, dir, dir ? HH : 0);
        }
    }

    // attention
    gemm_bf16_wmma<<<dim3(8, 256), 256, 0, stream>>>(
        HSEQB, S1WB, Z, (int)MBT, DD, DD, DD, DD, DD, s1_b, nullptr, 0, nullptr, 0);
    colstats<<<dim3(DD), 256, 0, stream>>>(Z, (int)MBT, DD, MEANB, VARB);
    alpha_kernel<<<dim3((int)MBT), 256, 0, stream>>>(Z, MEANB, VARB, s2_w, ALPHA);
    att_pool<<<dim3(BB), 256, 0, stream>>>(HSEQ, ALPHA, slen, UB);

    // phase 2
    chunk_att<<<dim3(NE * 8), 256, 0, stream>>>(UB, CHUNK);
    build_delta2<<<dim3(NE * 8), 256, 0, stream>>>(UB, CHUNK, pos, D2);
    gemm_bf16_wmma<<<dim3(8, 4), 256, 0, stream>>>(
        D2, W2WB, HID2, NE * 8, DD, K2P, K2P, K2P, DD, W2_b, nullptr, 0, nullptr, 0);
    bn_leaky_group<<<dim3(NE), 256, 0, stream>>>(HID2, 8);
    head2_kernel<<<dim3(NE * 8), 256, 0, stream>>>(HID2, Wo_w, Wo_b, out, EXTRA);

    // phase 3 (D3/HID3 reuse XG region; Z is dead here)
    build_delta3<<<dim3(NE * NCC), 256, 0, stream>>>(UB, dis, EXTRA, D3);
    gemm_bf16_wmma<<<dim3(8, 64), 256, 0, stream>>>(
        D3, W3WB, HID3, NE * NCC, DD, K3P, K3P, K3P, DD, W3_b, nullptr, 0, nullptr, 0);
    bn_leaky_group<<<dim3(NE), 256, 0, stream>>>(HID3, NCC);
    head2_kernel<<<dim3(NE * NCC), 256, 0, stream>>>(
        HID3, cls_w, cls_b, out + NE * 8 * 2, nullptr);

    // L
    float* Lout = out + NE * 8 * 2 + NE * NCC * 2;
    zero_f<<<dim3((NE * NCC + 255) / 256), 256, 0, stream>>>(Lout, (long)NE * NCC);
    setL<<<dim3(1), 64, 0, stream>>>(Lout, lab3);
}
